// RNNModel_22668837388965
// MI455X (gfx1250) — matrix operand order repaired and checked by reading, not compiled, not run on hardware
//
#include <hip/hip_runtime.h>

// ---------------------------------------------------------------------------
// 3-layer LSTM forward (AWD-LSTM eval) for MI455X / gfx1250.
//   - bf16 WMMA (v_wmma_f32_16x16x32_bf16) for input projections and the
//     recurrent GEMM, fp32 accumulation, fp32 gate math.
//   - Weights converted to padded bf16 once per call (L2-resident thereafter).
//   - x_proj computed in time-chunks (big parallel GEMM), then one fused
//     step kernel per timestep: 4-gate GEMM tile + LSTM cell + state update.
// ---------------------------------------------------------------------------

typedef __attribute__((ext_vector_type(16))) __bf16 v16bf;
typedef __attribute__((ext_vector_type(8)))  __bf16 v8bf;
typedef __attribute__((ext_vector_type(8)))  float  v8f;

#define S_TOT 200
#define BATCH 128
#define NINP  400
#define NHID  1150
#define HP    1152      // NHID padded to multiple of 32 (K) / 16 (N tiles)
#define G4    4608      // 4*HP rows of gate weights (rows 4600..4607 zero)
#define KP0   416       // NINP padded to multiple of 32

// ---- WMMA tile loaders (contiguous k = 16*(lane>>4) + s, same order for A and B) ------
// A (16x32 bf16): lanes 0-15 -> M=lane, K = 0..15; lanes 16-31 ->
// M=lane-16, K = 16..31. Two 16-byte contiguous loads per lane.
__device__ __forceinline__ v16bf load_a_tile(const __bf16* __restrict__ base,
                                             int ld, int row0, int k0, int lane) {
  const int r   = lane & 15;
  const int off = (lane >> 4) * 16;
  const __bf16* p = base + (size_t)(row0 + r) * ld + (k0 + off);
  v8bf lo = *(const v8bf*)p;
  v8bf hi = *(const v8bf*)(p + 8);
  v16bf a;
#pragma unroll
  for (int i = 0; i < 8; ++i) { a[i] = lo[i]; a[i + 8] = hi[i]; }
  return a;
}

// B (32x16 bf16), B[k][n] = W[n][k]: lane column c = lane&15 reads W row
// (nrow0+c), K range (lane>>4)*16 .. +15 -> one contiguous 32-byte load.
__device__ __forceinline__ v16bf load_b_tile(const __bf16* __restrict__ wbase,
                                             int ld, int nrow0, int k0, int lane) {
  const int c  = lane & 15;
  const int kh = (lane >> 4) * 16;
  return *(const v16bf*)(wbase + (size_t)(nrow0 + c) * ld + (k0 + kh));
}

__device__ __forceinline__ float sigmoidf_(float x) {
  return 1.0f / (1.0f + __expf(-x));
}

// ---- small prep kernels ----------------------------------------------------
__global__ void conv_weight_kernel(const float* __restrict__ W, __bf16* __restrict__ Wb,
                                   int nrows, int K, int Kp, size_t total) {
  for (size_t i = (size_t)blockIdx.x * blockDim.x + threadIdx.x; i < total;
       i += (size_t)gridDim.x * blockDim.x) {
    int r = (int)(i / Kp), k = (int)(i % Kp);
    float v = (r < nrows && k < K) ? W[(size_t)r * K + k] : 0.0f;
    Wb[i] = (__bf16)v;
  }
}

__global__ void bias_kernel(const float* __restrict__ bi, const float* __restrict__ bh,
                            float* __restrict__ out) {
  int n = blockIdx.x * blockDim.x + threadIdx.x;
  if (n < G4) out[n] = (n < 4 * NHID) ? (bi[n] + bh[n]) : 0.0f;
}

__global__ void embed_kernel(const int* __restrict__ tokens, const float* __restrict__ emb,
                             __bf16* __restrict__ x0) {
  const size_t total = (size_t)S_TOT * BATCH * KP0;
  for (size_t i = (size_t)blockIdx.x * blockDim.x + threadIdx.x; i < total;
       i += (size_t)gridDim.x * blockDim.x) {
    int row = (int)(i / KP0), k = (int)(i % KP0);
    float v = (k < NINP) ? emb[(size_t)tokens[row] * NINP + k] : 0.0f;
    x0[i] = (__bf16)v;
  }
}

__global__ void init_state_kernel(const float* __restrict__ h0, const float* __restrict__ c0,
                                  __bf16* __restrict__ hA, __bf16* __restrict__ hB,
                                  float* __restrict__ cst) {
  const int total = BATCH * HP;
  for (int i = blockIdx.x * blockDim.x + threadIdx.x; i < total;
       i += gridDim.x * blockDim.x) {
    int b = i / HP, j = i % HP;
    float hv = (j < NHID) ? h0[(size_t)b * NHID + j] : 0.0f;
    float cv = (j < NHID) ? c0[(size_t)b * NHID + j] : 0.0f;
    hA[i] = (__bf16)hv;
    hB[i] = (__bf16)hv;
    cst[i] = cv;
  }
}

// ---- input projection GEMM: XP[rows][4608] = X[rows][Kp] * Wih^T + bias ----
// One wave per block: 1 M-tile x 4 N-tiles (A fetch reused 4x).
__global__ __launch_bounds__(32) void proj_gemm_kernel(
    const __bf16* __restrict__ X, const __bf16* __restrict__ W,
    const float* __restrict__ bias, float* __restrict__ XP, int Kp) {
  const int nt4  = blockIdx.x;                 // 0..71 -> 4 n-tiles of 16
  const int mt   = blockIdx.y;
  const int lane = threadIdx.x;
  const int c16  = lane & 15;
  const int m0   = mt * 16 + (lane >> 4) * 8;

  v8f acc[4];
#pragma unroll
  for (int j = 0; j < 4; ++j) {
    float bv = bias[nt4 * 64 + j * 16 + c16];
#pragma unroll
    for (int r = 0; r < 8; ++r) acc[j][r] = bv;
  }

  for (int k0 = 0; k0 < Kp; k0 += 32) {
    v16bf a = load_a_tile(X, Kp, mt * 16, k0, lane);
#pragma unroll
    for (int j = 0; j < 4; ++j) {
      v16bf b = load_b_tile(W, Kp, nt4 * 64 + j * 16, k0, lane);
      acc[j] = __builtin_amdgcn_wmma_f32_16x16x32_bf16(
          false, a, false, b, (short)0, acc[j], false, false);
    }
  }

#pragma unroll
  for (int j = 0; j < 4; ++j) {
    int n = nt4 * 64 + j * 16 + c16;
#pragma unroll
    for (int r = 0; r < 8; ++r)
      XP[(size_t)(m0 + r) * G4 + n] = acc[j][r];
  }
}

// ---- fused LSTM step: gates = xp + h_prev * Whh^T, then cell update --------
// grid = (HP/16 hidden-col tiles, BATCH/16 row tiles), 1 wave per block.
// Each wave computes the i,f,g,o gate tiles for its (rows, hidden cols).
__global__ __launch_bounds__(32) void lstm_step_kernel(
    const float* __restrict__ xp,        // [BATCH][4608] (bias folded in)
    const __bf16* __restrict__ hprev,    // [BATCH][HP]
    __bf16* __restrict__ hnext,          // [BATCH][HP]
    float* __restrict__ cstate,          // [BATCH][HP]
    const __bf16* __restrict__ Whh,      // [4608][HP] bf16
    __bf16* __restrict__ seq_out,        // [BATCH][HP] slice or null
    float* __restrict__ y_out,           // [BATCH][NHID] slice or null
    float* __restrict__ hT_out,          // [BATCH][NHID] or null (t==S-1)
    float* __restrict__ cT_out) {
  const int ct   = blockIdx.x;           // hidden col tile 0..71
  const int rt   = blockIdx.y;           // batch row tile 0..7
  const int lane = threadIdx.x;
  const int col  = ct * 16 + (lane & 15);
  const int m0   = rt * 16 + (lane >> 4) * 8;

  v8f acc[4];
#pragma unroll
  for (int g = 0; g < 4; ++g) {
    int n = g * NHID + col;              // gate bases are NOT 16-aligned: fine,
#pragma unroll                           // max index 3*1150+1151 = 4601 < 4608
    for (int r = 0; r < 8; ++r) acc[g][r] = xp[(size_t)(m0 + r) * G4 + n];
  }

  for (int k0 = 0; k0 < HP; k0 += 32) {
    v16bf a = load_a_tile(hprev, HP, rt * 16, k0, lane);
#pragma unroll
    for (int g = 0; g < 4; ++g) {
      v16bf b = load_b_tile(Whh, HP, g * NHID + ct * 16, k0, lane);
      acc[g] = __builtin_amdgcn_wmma_f32_16x16x32_bf16(
          false, a, false, b, (short)0, acc[g], false, false);
    }
  }

  if (col < NHID) {
#pragma unroll
    for (int r = 0; r < 8; ++r) {
      const int m = m0 + r;
      float ig = sigmoidf_(acc[0][r]);
      float fg = sigmoidf_(acc[1][r]);
      float gg = tanhf(acc[2][r]);
      float og = sigmoidf_(acc[3][r]);
      float cold = cstate[(size_t)m * HP + col];
      float cnew = fg * cold + ig * gg;
      float h    = og * tanhf(cnew);
      cstate[(size_t)m * HP + col] = cnew;
      hnext[(size_t)m * HP + col]  = (__bf16)h;
      if (seq_out) seq_out[(size_t)m * HP + col] = (__bf16)h;
      if (y_out)   y_out[(size_t)m * NHID + col] = h;
      if (hT_out) {
        hT_out[(size_t)m * NHID + col] = h;
        cT_out[(size_t)m * NHID + col] = cnew;
      }
    }
  }
}

// ---------------------------------------------------------------------------
extern "C" void kernel_launch(void* const* d_in, const int* in_sizes, int n_in,
                              void* d_out, int out_size, void* d_ws, size_t ws_size,
                              hipStream_t stream) {
  (void)in_sizes; (void)n_in; (void)out_size;
  const int*   tokens = (const int*)d_in[0];
  const float* h0     = (const float*)d_in[1];
  const float* c0     = (const float*)d_in[2];
  const float* emb    = (const float*)d_in[3];
  const float *W_ih[3], *W_hh[3], *b_ih[3], *b_hh[3];
  for (int l = 0; l < 3; ++l) {
    W_ih[l] = (const float*)d_in[4 + 4 * l + 0];
    W_hh[l] = (const float*)d_in[4 + 4 * l + 1];
    b_ih[l] = (const float*)d_in[4 + 4 * l + 2];
    b_hh[l] = (const float*)d_in[4 + 4 * l + 3];
  }

  // ---- carve workspace ----
  char* ws = (char*)d_ws;
  size_t off = 0;
  auto carve = [&](size_t bytes) -> char* {
    char* p = ws + off;
    off = (off + bytes + 255) & ~(size_t)255;
    return p;
  };
  __bf16* x0   = (__bf16*)carve((size_t)S_TOT * BATCH * KP0 * 2);
  __bf16* seqA = (__bf16*)carve((size_t)S_TOT * BATCH * HP * 2);
  __bf16* seqB = (__bf16*)carve((size_t)S_TOT * BATCH * HP * 2);
  __bf16 *Wihb[3], *Whhb[3];
  float* biasb[3];
  __bf16* hbuf[3][2];
  float* cst[3];
  for (int l = 0; l < 3; ++l) {
    int Kp = (l == 0) ? KP0 : HP;
    Wihb[l]  = (__bf16*)carve((size_t)G4 * Kp * 2);
    Whhb[l]  = (__bf16*)carve((size_t)G4 * HP * 2);
    biasb[l] = (float*)carve((size_t)G4 * 4);
    hbuf[l][0] = (__bf16*)carve((size_t)BATCH * HP * 2);
    hbuf[l][1] = (__bf16*)carve((size_t)BATCH * HP * 2);
    cst[l]     = (float*)carve((size_t)BATCH * HP * 4);
  }
  // x_proj chunk buffer: adapt chunk length to remaining workspace
  size_t avail = (ws_size > off) ? (ws_size - off) : 0;
  const size_t xp_step = (size_t)BATCH * G4 * 4;
  int S_c = (int)(avail / xp_step);
  if (S_c < 1)  S_c = 1;
  if (S_c > 50) S_c = 50;
  float* xpbuf = (float*)(ws + off);

  // ---- prep ----
  for (int l = 0; l < 3; ++l) {
    int K = (l == 0) ? NINP : NHID, Kp = (l == 0) ? KP0 : HP;
    size_t ni = (size_t)G4 * Kp, nh = (size_t)G4 * HP;
    conv_weight_kernel<<<4096, 256, 0, stream>>>(W_ih[l], Wihb[l], 4 * NHID, K, Kp, ni);
    conv_weight_kernel<<<4096, 256, 0, stream>>>(W_hh[l], Whhb[l], 4 * NHID, NHID, HP, nh);
    bias_kernel<<<(G4 + 255) / 256, 256, 0, stream>>>(b_ih[l], b_hh[l], biasb[l]);
    init_state_kernel<<<(BATCH * HP + 255) / 256, 256, 0, stream>>>(
        h0 + (size_t)l * BATCH * NHID, c0 + (size_t)l * BATCH * NHID,
        hbuf[l][0], hbuf[l][1], cst[l]);
  }
  embed_kernel<<<4096, 256, 0, stream>>>(tokens, emb, x0);
  hipMemsetAsync(seqA, 0, (size_t)S_TOT * BATCH * HP * 2, stream);  // zero pad cols
  hipMemsetAsync(seqB, 0, (size_t)S_TOT * BATCH * HP * 2, stream);

  float* out_res = (float*)d_out;
  float* out_hT  = out_res + (size_t)S_TOT * BATCH * NHID;
  float* out_cT  = out_hT + (size_t)3 * BATCH * NHID;

  // ---- layers ----
  for (int l = 0; l < 3; ++l) {
    const __bf16* xin = (l == 0) ? x0 : ((l == 1) ? seqA : seqB);
    const int Kp = (l == 0) ? KP0 : HP;
    __bf16* seq_out = (l == 0) ? seqA : ((l == 1) ? seqB : nullptr);

    for (int s0 = 0; s0 < S_TOT; s0 += S_c) {
      const int sc = (S_TOT - s0 < S_c) ? (S_TOT - s0) : S_c;
      const int rows = sc * BATCH;
      dim3 pg(G4 / 64, rows / 16);
      proj_gemm_kernel<<<pg, 32, 0, stream>>>(
          xin + (size_t)s0 * BATCH * Kp, Wihb[l], biasb[l], xpbuf, Kp);

      for (int ti = 0; ti < sc; ++ti) {
        const int t = s0 + ti;
        const float* xpt = xpbuf + (size_t)ti * BATCH * G4;
        __bf16* so = seq_out ? seq_out + (size_t)t * BATCH * HP : nullptr;
        float* yo  = (l == 2) ? out_res + (size_t)t * BATCH * NHID : nullptr;
        float* hT  = (t == S_TOT - 1) ? out_hT + (size_t)l * BATCH * NHID : nullptr;
        float* cT  = (t == S_TOT - 1) ? out_cT + (size_t)l * BATCH * NHID : nullptr;
        dim3 sg(HP / 16, BATCH / 16);
        lstm_step_kernel<<<sg, 32, 0, stream>>>(
            xpt, hbuf[l][t & 1], hbuf[l][(t & 1) ^ 1], cst[l], Whhb[l],
            so, yo, hT, cT);
      }
    }
  }
}
